// GAB_16647293240139
// MI455X (gfx1250) — compile-verified, matrix operand order checked by reading, not run on hardware
//
#include <hip/hip_runtime.h>
#include <hip/hip_bf16.h>
#include <cstdint>

typedef __attribute__((ext_vector_type(16))) __bf16 v16bf;
typedef __attribute__((ext_vector_type(8)))  float  v8f;

#define DEV __device__ __forceinline__

// Problem sizes (fixed by the reference)
constexpr int Bb  = 4;
constexpr int Ls  = 2048;
constexpr int Dd  = 1024;
constexpr int NHh = 16;
constexpr int MBm = 16;
constexpr int Kc  = 4;
constexpr int HDd = Dd / NHh;   // 64
constexpr int NMBn = Ls / MBm;  // 128
constexpr int ML  = Bb * Ls;    // 8192 rows for the big GEMMs

// ---------------- CDNA5 feature gates (device pass only) ----------------
#if defined(__HIP_DEVICE_COMPILE__) && __has_builtin(__builtin_amdgcn_tensor_load_to_lds) && \
    __has_builtin(__builtin_amdgcn_s_wait_tensorcnt)
#define HAS_TDM 1
#endif

// ---------------- math helpers ----------------
DEV float fast_tanh(float x) {
#if defined(__HIP_DEVICE_COMPILE__) && __has_builtin(__builtin_amdgcn_tanhf)
  return __builtin_amdgcn_tanhf(x);   // v_tanh_f32 (CDNA5 TRANS op)
#elif defined(__HIP_DEVICE_COMPILE__) && __has_builtin(__builtin_amdgcn_tanh_f32)
  return __builtin_amdgcn_tanh_f32(x);
#else
  return tanhf(x);
#endif
}
DEV float gelu_t(float x) {
  float t = fast_tanh(0.79788456f * x * (1.0f + 0.044715f * x * x));
  return 0.5f * x * (1.0f + t);
}
DEV float gelu_bw(float x) {
  float t = fast_tanh(0.79788456f * x * (1.0f + 0.044715f * x * x));
  return 0.5f * x * ((1.0f - t * t) * (0.79788456f + 0.1070322243f * x * x)) + 0.5f * (1.0f + t);
}
DEV float sigm(float x) { return 1.0f / (1.0f + __expf(-x)); }

// sum across a 16-lane group (rows mapped to lane groups; wave32 => 2 groups/wave)
DEV float red16(float v) {
  v += __shfl_xor(v, 1, 16);
  v += __shfl_xor(v, 2, 16);
  v += __shfl_xor(v, 4, 16);
  v += __shfl_xor(v, 8, 16);
  return v;
}

// ---------------- Tensor Data Mover: 2D tile global -> LDS ----------------
#if defined(HAS_TDM)
typedef unsigned int v4u __attribute__((ext_vector_type(4)));
typedef int          t8i __attribute__((ext_vector_type(8)));
typedef int          t4i __attribute__((ext_vector_type(4)));
// D# group0/group1 per CDNA5 ISA 8.3/8.4. data_size=1 (2-byte elems), 2D tile,
// dims/strides in elements. Tile lands contiguously row-major in LDS.
DEV void tdm_load_2d(void* lds, const void* g, unsigned tensor_d0, unsigned tensor_d1,
                     unsigned tile_d0, unsigned tile_d1, unsigned long long stride0) {
  const unsigned long long ga = (unsigned long long)(uintptr_t)g;
  v4u g0;
  g0[0] = 1u;                                  // count=1 (valid descriptor)
  g0[1] = (unsigned)(uintptr_t)lds;            // lds_addr (low 32 bits of generic ptr)
  g0[2] = (unsigned)ga;                        // global_addr[31:0]
  g0[3] = (unsigned)((ga >> 32) & 0x01FFFFFFu) | (2u << 30);  // addr[56:32] | type=2
  t8i g1;
  g1[0] = (int)(1u << 16);                                    // data_size = 1 (2 bytes)
  g1[1] = (int)((tensor_d0 & 0xFFFFu) << 16);                 // tensor_dim0[15:0]
  g1[2] = (int)((tensor_d0 >> 16) | ((tensor_d1 & 0xFFFFu) << 16));
  g1[3] = (int)((tensor_d1 >> 16) | (tile_d0 << 16));         // tile_dim0
  g1[4] = (int)(tile_d1 & 0xFFFFu);                           // tile_dim1 (tile_dim2=0)
  g1[5] = (int)(unsigned)stride0;                             // tensor_dim0_stride lo
  g1[6] = (int)(unsigned)((stride0 >> 32) & 0xFFFFu);         // stride hi (dim1_stride=0)
  g1[7] = 0;
  t4i z4 = {0, 0, 0, 0};
#if defined(__clang_major__) && (__clang_major__ >= 23)
  t8i z8 = {0, 0, 0, 0, 0, 0, 0, 0};
  __builtin_amdgcn_tensor_load_to_lds(g0, g1, z4, z4, z8, 0);
#else
  __builtin_amdgcn_tensor_load_to_lds(g0, g1, z4, z4, 0);
#endif
}
#endif

// ---------------- WMMA core (v_wmma_f32_16x16x32_bf16) ----------------
// A fragment (16x32 bf16, ISA 7.12.2): lanes 0-15 -> M=lane, K pairs {0..7,16..23};
// lanes 16-31 -> M=lane-16, K pairs {8..15,24..31}. B symmetric with N<->M.
template <typename LA, typename LB>
DEV v8f wmma_acc_tile(v8f acc, int K, LA la, LB lb, int lane) {
  const int half = (lane >> 4) & 1;
  const int l = lane & 15;
  for (int kk = 0; kk < K; kk += 32) {
    v16bf a, b;
#pragma unroll
    for (int j = 0; j < 16; ++j) {
      const int vg = j >> 1;
      const int kb = (vg < 4) ? (vg * 2) : (16 + (vg - 4) * 2);
      const int k = kk + kb + (j & 1) + (half ? 8 : 0);
      a[j] = la(l, k);
      b[j] = lb(l, k);
    }
    acc = __builtin_amdgcn_wmma_f32_16x16x32_bf16(false, a, false, b, (short)0, acc,
                                                  false, false);
  }
  return acc;
}

// C/D layout: VGPR v, lanes 0-15 -> (row v, col lane); lanes 16-31 -> (row v+8, col lane-16)
template <typename SC>
DEV void store_tile(const v8f& acc, SC sc, int lane) {
  const int half = lane >> 4;
  const int l = lane & 15;
#pragma unroll
  for (int v = 0; v < 8; ++v) sc(v + half * 8, l, acc[v]);
}

// Generic LDS-resident GEMM distributed over 8 wave32s (blockDim == 256).
template <typename LA, typename LB, typename SC>
DEV void gemm_lds(int M, int N, int K, LA la, LB lb, SC sc, int wave, int lane) {
  const int mt_n = M >> 4, nt_n = N >> 4;
  for (int t = wave; t < mt_n * nt_n; t += 8) {
    const int mt = t / nt_n, nt = t % nt_n;
    v8f acc;
#pragma unroll
    for (int j = 0; j < 8; ++j) acc[j] = 0.0f;
    acc = wmma_acc_tile(
        acc, K, [&](int m, int k) { return la(mt * 16 + m, k); },
        [&](int n, int k) { return lb(nt * 16 + n, k); }, lane);
    store_tile(acc, [&](int m, int n, float v) { sc(mt * 16 + m, nt * 16 + n, v); }, lane);
  }
}

// ---------------- kernel: f32 -> bf16 convert ----------------
__global__ void k_cvt(const float* __restrict__ in, __bf16* __restrict__ out, int n) {
  for (int i = blockIdx.x * blockDim.x + threadIdx.x; i < n; i += gridDim.x * blockDim.x)
    out[i] = (__bf16)in[i];
}

// ---------------- kernel: fused f32->bf16 convert + transpose (for GEMM B operands) ----
// out[c][r] = (bf16) in[r][c], square D x D, 32x32 LDS tiles.
__global__ __launch_bounds__(256) void k_cvt_t(const float* __restrict__ in,
                                               __bf16* __restrict__ out) {
  __shared__ float tile[32][33];
  const int nbx = Dd / 32;
  const int bx = blockIdx.x % nbx, by = blockIdx.x / nbx;
  const int tx = threadIdx.x & 31, ty = threadIdx.x >> 5;  // 32 x 8
#pragma unroll
  for (int i = 0; i < 32; i += 8)
    tile[ty + i][tx] = in[(size_t)(by * 32 + ty + i) * Dd + bx * 32 + tx];
  __syncthreads();
#pragma unroll
  for (int i = 0; i < 32; i += 8)
    out[(size_t)(bx * 32 + ty + i) * Dd + by * 32 + tx] = (__bf16)tile[tx][ty + i];
}

// ---------------- kernel: big bf16 WMMA GEMM  C(f32) = A(bf16) @ Bt(bf16, N-major) ------
constexpr int BM = 128, BN = 64, BK = 32;
__global__ __launch_bounds__(256) void k_gemm(const __bf16* __restrict__ A,
                                              const __bf16* __restrict__ Bt,
                                              float* __restrict__ C, int M, int N, int Kk) {
  __shared__ alignas(16) __bf16 sA[BM * BK];   // [BM][BK] row-major
  __shared__ alignas(16) __bf16 sBt[BN * BK];  // [BN][BK] (B already transposed in global)
  const int tid = threadIdx.x, lane = tid & 31, wave = tid >> 5;
  const int nb = N / BN;
  const int m0 = (blockIdx.x / nb) * BM;
  const int n0 = (blockIdx.x % nb) * BN;

  v8f acc[4];
#pragma unroll
  for (int i = 0; i < 4; ++i)
#pragma unroll
    for (int j = 0; j < 8; ++j) acc[i][j] = 0.0f;

  for (int kk = 0; kk < Kk; kk += BK) {
#if defined(HAS_TDM)
    // Tensor Data Mover DMA: both tiles are contiguous 2D sub-tiles.
    if (wave == 0) {
      tdm_load_2d(sA, A + (size_t)m0 * Kk + kk, (unsigned)Kk, (unsigned)M, BK, BM,
                  (unsigned long long)Kk);
      tdm_load_2d(sBt, Bt + (size_t)n0 * Kk + kk, (unsigned)Kk, (unsigned)N, BK, BN,
                  (unsigned long long)Kk);
      __builtin_amdgcn_s_wait_tensorcnt(0);
    }
    __syncthreads();
#else
    {  // vectorized fallback staging: 16B chunks
      const uint4* Ag = (const uint4*)(A + (size_t)m0 * Kk + kk);
      const uint4* Bg = (const uint4*)(Bt + (size_t)n0 * Kk + kk);
      const int pitch = Kk >> 3;  // uint4 per row
#pragma unroll
      for (int i = 0; i < 2; ++i) {  // 512 chunks of A
        const int idx = tid + 256 * i;
        ((uint4*)sA)[idx] = Ag[(size_t)(idx >> 2) * pitch + (idx & 3)];
      }
      ((uint4*)sBt)[tid] = Bg[(size_t)(tid >> 2) * pitch + (tid & 3)];  // 256 chunks of B
    }
    __syncthreads();
#endif
    if (kk + BK < Kk) {  // global_prefetch_b8 on next K-slab
      __builtin_prefetch(&A[(size_t)(m0 + (tid & 127)) * Kk + kk + BK], 0, 1);
      __builtin_prefetch(&Bt[(size_t)(n0 + (tid & 63)) * Kk + kk + BK], 0, 1);
    }
#pragma unroll
    for (int i = 0; i < 4; ++i) {  // 8x4 tiles, 4 per wave
      const int id = wave + i * 8;
      const int mt = id >> 2, nt = id & 3;
      acc[i] = wmma_acc_tile(
          acc[i], BK, [&](int m, int k) { return sA[(mt * 16 + m) * BK + k]; },
          [&](int n, int k) { return sBt[(nt * 16 + n) * BK + k]; }, lane);
    }
    __syncthreads();
  }
#pragma unroll
  for (int i = 0; i < 4; ++i) {
    const int id = wave + i * 8;
    const int mt = id >> 2, nt = id & 3;
    store_tile(acc[i],
               [&](int m, int n, float v) {
                 C[(size_t)(m0 + mt * 16 + m) * N + n0 + nt * 16 + n] = v;
               },
               lane);
  }
}

// ---------------- kernel: causal depthwise conv (K=4), both q and k filters ----------------
__global__ __launch_bounds__(256) void k_dwconv(const float* __restrict__ xq,
                                                const float* __restrict__ wq,
                                                const float* __restrict__ bq,
                                                const float* __restrict__ wk,
                                                const float* __restrict__ bk,
                                                float* __restrict__ XQ,
                                                float* __restrict__ XK) {
  const int bl = blockIdx.x;
  const int b = bl / Ls, l = bl % Ls;
  for (int d = threadIdx.x; d < Dd; d += 256) {
    float aq = bq[d], ak = bk[d];
#pragma unroll
    for (int j = 0; j < Kc; ++j) {
      const int ll = l - (Kc - 1) + j;
      if (ll >= 0) {
        const float xv = xq[((size_t)b * Ls + ll) * Dd + d];
        aq += xv * wq[d * Kc + j];
        ak += xv * wk[d * Kc + j];
      }
    }
    const size_t o = ((size_t)b * Ls + l) * Dd + d;
    XQ[o] = aq;
    XK[o] = ak;
  }
}

// ---------------- kernel: ttt_lr_eta = sigmoid(x . lr_w + lr_b) / HD ----------------
__global__ __launch_bounds__(256) void k_eta(const float* __restrict__ x,
                                             const float* __restrict__ lw,
                                             const float* __restrict__ lb,
                                             float* __restrict__ lre) {
  const int bl = blockIdx.x;
  const int b = bl / Ls, l = bl % Ls;
  const int h = threadIdx.x >> 4, g = threadIdx.x & 15;
  float s = 0.0f;
  for (int c = g; c < Dd; c += 16) s += x[((size_t)b * Ls + l) * Dd + c] * lw[h * Dd + c];
  s = red16(s);
  if (g == 0) lre[((size_t)b * NHh + h) * Ls + l] = sigm(s + lb[h]) * (1.0f / (float)HDd);
}

// ---------------- kernel: TTT scan, one block per (b,h), state resident in LDS ----------------
constexpr size_t SCAN_LDS =
    (size_t)64 * 256 * 4 + 256 * 64 * 4 + 256 * 4 + 64 * 4  // W1,W2,b1,b2
    + 3 * (16 * 64 * 4)                                     // Xq,Xk,Xv
    + 5 * (16 * 256 * 4)                                    // Z1,X2,gZ1,Z1b,X2b
    + 3 * (16 * 64 * 4)                                     // Z2,gZ2,Z2b
    + 2 * (16 * 16 * 4)                                     // C1,C2
    + 3 * 64;                                               // Eta,Tok,LastEta

__global__ __launch_bounds__(256) void k_scan(
    const float* __restrict__ XQ, const float* __restrict__ XK, const float* __restrict__ XV,
    const float* __restrict__ lre, const float* __restrict__ tokidx,
    const float* __restrict__ nw, const float* __restrict__ nb,
    const float* __restrict__ W1g, const float* __restrict__ b1g,
    const float* __restrict__ W2g, const float* __restrict__ b2g,
    float* __restrict__ tout) {
  extern __shared__ char smem[];
  char* p = smem;
  auto alloc = [&](size_t bytes) { char* r = p; p += bytes; return r; };
  float* sW1  = (float*)alloc(64 * 256 * 4);   // W1[k][n]  (64 x 256)
  float* sW2  = (float*)alloc(256 * 64 * 4);   // W2[k][n]  (256 x 64)
  float* sb1  = (float*)alloc(256 * 4);
  float* sb2  = (float*)alloc(64 * 4);
  float* sXq  = (float*)alloc(16 * 64 * 4);
  float* sXk  = (float*)alloc(16 * 64 * 4);
  float* sXv  = (float*)alloc(16 * 64 * 4);
  float* sZ1  = (float*)alloc(16 * 256 * 4);
  float* sX2  = (float*)alloc(16 * 256 * 4);
  float* sgZ1 = (float*)alloc(16 * 256 * 4);
  float* sZ1b = (float*)alloc(16 * 256 * 4);
  float* sX2b = (float*)alloc(16 * 256 * 4);
  float* sZ2  = (float*)alloc(16 * 64 * 4);
  float* sgZ2 = (float*)alloc(16 * 64 * 4);
  float* sZ2b = (float*)alloc(16 * 64 * 4);
  float* sC1  = (float*)alloc(16 * 16 * 4);
  float* sC2  = (float*)alloc(16 * 16 * 4);
  float* sEta = (float*)alloc(64);
  float* sTok = (float*)alloc(64);
  float* sLE  = (float*)alloc(64);

  const int tid = threadIdx.x, lane = tid & 31, wave = tid >> 5;
  const int b = blockIdx.x / NHh, h = blockIdx.x % NHh;

  for (int i = tid; i < 64 * 256; i += 256) sW1[i] = W1g[(size_t)h * 64 * 256 + i];
  for (int i = tid; i < 256 * 64; i += 256) sW2[i] = W2g[(size_t)h * 256 * 64 + i];
  sb1[tid] = b1g[h * 256 + tid];
  if (tid < 64) sb2[tid] = b2g[h * 64 + tid];
  if (tid < 16) sTok[tid] = fmaxf(1.0f / (float)(tid + 1) + tokidx[tid], 0.0f);
  __syncthreads();

  const float* gw = nw + h * HDd;
  const float* gb = nb + h * HDd;

  for (int n = 0; n < NMBn; ++n) {
    const size_t rowbase = ((size_t)b * Ls + (size_t)n * MBm) * Dd + (size_t)h * HDd;
    for (int i = tid; i < 16 * 64; i += 256) {
      const int r = i >> 6, c = i & 63;
      const size_t gidx = rowbase + (size_t)r * Dd + c;
      sXq[i] = XQ[gidx];
      sXk[i] = XK[gidx];
      sXv[i] = XV[gidx];
    }
    if (tid < 16) sEta[tid] = lre[((size_t)b * NHh + h) * Ls + n * MBm + tid];
    __syncthreads();
    if (tid < 16) sLE[tid] = sTok[15] * sEta[tid];  // last_eta[j] = tok[MB-1] * lr_eta[j]
    __syncthreads();

    // Z1 = XK @ W1 + b1            (16x256, K=64)
    gemm_lds(16, 256, 64,
             [&](int m, int k) { return (__bf16)sXk[m * 64 + k]; },
             [&](int nn, int k) { return (__bf16)sW1[k * 256 + nn]; },
             [&](int m, int nn, float v) { sZ1[m * 256 + nn] = v + sb1[nn]; }, wave, lane);
    __syncthreads();
    for (int i = tid; i < 16 * 256; i += 256) sX2[i] = gelu_t(sZ1[i]);
    __syncthreads();

    // Z2 = X2 @ W2 + b2            (16x64, K=256)
    gemm_lds(16, 64, 256,
             [&](int m, int k) { return (__bf16)sX2[m * 256 + k]; },
             [&](int nn, int k) { return (__bf16)sW2[k * 64 + nn]; },
             [&](int m, int nn, float v) { sZ2[m * 64 + nn] = v + sb2[nn]; }, wave, lane);
    __syncthreads();

    // gZ2 = ln_fused_l2_bwd(Z2, XV - XK, gw, gb)
    {
      const int r = tid >> 4, g16 = tid & 15;
      float x4[4], s = 0.0f, sq = 0.0f;
#pragma unroll
      for (int i = 0; i < 4; ++i) {
        const float v = sZ2[r * 64 + g16 + 16 * i];
        x4[i] = v; s += v; sq += v * v;
      }
      s = red16(s); sq = red16(sq);
      const float mu = s * (1.0f / 64.0f);
      const float var = sq * (1.0f / 64.0f) - mu * mu;
      const float rstd = rsqrtf(var + 1e-6f);
      float gx4[4], xh4[4], sg = 0.0f, sgx = 0.0f;
#pragma unroll
      for (int i = 0; i < 4; ++i) {
        const int c = g16 + 16 * i;
        const float xh = (x4[i] - mu) * rstd; xh4[i] = xh;
        const float tgt = sXv[r * 64 + c] - sXk[r * 64 + c];
        const float gxv = (gw[c] * xh + gb[c] - tgt) * gw[c];
        gx4[i] = gxv; sg += gxv; sgx += gxv * xh;
      }
      sg = red16(sg); sgx = red16(sgx);
#pragma unroll
      for (int i = 0; i < 4; ++i) {
        const int c = g16 + 16 * i;
        sgZ2[r * 64 + c] = (64.0f * gx4[i] - sg - xh4[i] * sgx) * (rstd * (1.0f / 64.0f));
      }
    }
    __syncthreads();

    // gZ1 = (gZ2 @ W2^T) * gelu_bwd(Z1)   (16x256, K=64); B(k,n)=W2[n][k] contiguous
    gemm_lds(16, 256, 64,
             [&](int m, int k) { return (__bf16)sgZ2[m * 64 + k]; },
             [&](int nn, int k) { return (__bf16)sW2[nn * 64 + k]; },
             [&](int m, int nn, float v) { sgZ1[m * 256 + nn] = v * gelu_bw(sZ1[m * 256 + nn]); },
             wave, lane);
    __syncthreads();

    // C1 = eta .* (tril(XQ@XK^T) + tril)   (16x16, K=64)
    gemm_lds(16, 16, 64,
             [&](int m, int k) { return (__bf16)sXq[m * 64 + k]; },
             [&](int nn, int k) { return (__bf16)sXk[nn * 64 + k]; },
             [&](int m, int nn, float v) {
               sC1[m * 16 + nn] = (nn <= m) ? sTok[m] * sEta[nn] * (v + 1.0f) : 0.0f;
             },
             wave, lane);
    __syncthreads();

    // Z1_bar = XQ @ W1 + b1 - C1 @ gZ1
    gemm_lds(16, 256, 64,
             [&](int m, int k) { return (__bf16)sXq[m * 64 + k]; },
             [&](int nn, int k) { return (__bf16)sW1[k * 256 + nn]; },
             [&](int m, int nn, float v) { sZ1b[m * 256 + nn] = v + sb1[nn]; }, wave, lane);
    __syncthreads();
    gemm_lds(16, 256, 16,
             [&](int m, int k) { return (__bf16)(k < 16 ? sC1[m * 16 + k] : 0.0f); },
             [&](int nn, int k) { return (__bf16)(k < 16 ? sgZ1[k * 256 + nn] : 0.0f); },
             [&](int m, int nn, float v) { sZ1b[m * 256 + nn] -= v; }, wave, lane);
    __syncthreads();
    for (int i = tid; i < 16 * 256; i += 256) sX2b[i] = gelu_t(sZ1b[i]);
    __syncthreads();

    // C2 = eta .* (tril(X2b@X2^T) + tril)   (16x16, K=256)
    gemm_lds(16, 16, 256,
             [&](int m, int k) { return (__bf16)sX2b[m * 256 + k]; },
             [&](int nn, int k) { return (__bf16)sX2[nn * 256 + k]; },
             [&](int m, int nn, float v) {
               sC2[m * 16 + nn] = (nn <= m) ? sTok[m] * sEta[nn] * (v + 1.0f) : 0.0f;
             },
             wave, lane);
    __syncthreads();

    // Z2_bar = X2b @ W2 + b2 - C2 @ gZ2
    gemm_lds(16, 64, 256,
             [&](int m, int k) { return (__bf16)sX2b[m * 256 + k]; },
             [&](int nn, int k) { return (__bf16)sW2[k * 64 + nn]; },
             [&](int m, int nn, float v) { sZ2b[m * 64 + nn] = v + sb2[nn]; }, wave, lane);
    __syncthreads();
    gemm_lds(16, 64, 16,
             [&](int m, int k) { return (__bf16)(k < 16 ? sC2[m * 16 + k] : 0.0f); },
             [&](int nn, int k) { return (__bf16)(k < 16 ? sgZ2[k * 64 + nn] : 0.0f); },
             [&](int m, int nn, float v) { sZ2b[m * 64 + nn] -= v; }, wave, lane);
    __syncthreads();

    // out = XQ + ln_fwd(Z2_bar)
    {
      const int r = tid >> 4, g16 = tid & 15;
      float x4[4], s = 0.0f, sq = 0.0f;
#pragma unroll
      for (int i = 0; i < 4; ++i) {
        const float v = sZ2b[r * 64 + g16 + 16 * i];
        x4[i] = v; s += v; sq += v * v;
      }
      s = red16(s); sq = red16(sq);
      const float mu = s * (1.0f / 64.0f);
      const float var = sq * (1.0f / 64.0f) - mu * mu;
      const float rstd = rsqrtf(var + 1e-6f);
#pragma unroll
      for (int i = 0; i < 4; ++i) {
        const int c = g16 + 16 * i;
        const float xh = (x4[i] - mu) * rstd;
        tout[rowbase + (size_t)r * Dd + c] = sXq[r * 64 + c] + gw[c] * xh + gb[c];
      }
    }
    __syncthreads();

    // State updates (rank-16): W1 -= (lastEta*X1)^T @ gZ1 ; W2 -= (lastEta*X2)^T @ gZ2
    gemm_lds(64, 256, 16,
             [&](int m, int k) { return (__bf16)(k < 16 ? sLE[k] * sXk[k * 64 + m] : 0.0f); },
             [&](int nn, int k) { return (__bf16)(k < 16 ? sgZ1[k * 256 + nn] : 0.0f); },
             [&](int m, int nn, float v) { sW1[m * 256 + nn] -= v; }, wave, lane);
    gemm_lds(256, 64, 16,
             [&](int m, int k) { return (__bf16)(k < 16 ? sLE[k] * sX2[k * 256 + m] : 0.0f); },
             [&](int nn, int k) { return (__bf16)(k < 16 ? sgZ2[k * 64 + nn] : 0.0f); },
             [&](int m, int nn, float v) { sW2[m * 64 + nn] -= v; }, wave, lane);
    {
      float s = 0.0f;
      for (int k = 0; k < 16; ++k) s += sLE[k] * sgZ1[k * 256 + tid];
      sb1[tid] -= s;
      if (tid < 64) {
        float s2 = 0.0f;
        for (int k = 0; k < 16; ++k) s2 += sLE[k] * sgZ2[k * 64 + tid];
        sb2[tid] -= s2;
      }
    }
    __syncthreads();
  }
}

// ---------------- kernel: post-norm over D, gate with gelu(x@g_w), emit bf16 ----------------
__global__ __launch_bounds__(256) void k_postgate(const float* __restrict__ tout,
                                                  const float* __restrict__ xg,
                                                  const float* __restrict__ pw,
                                                  const float* __restrict__ pb,
                                                  __bf16* __restrict__ gbf) {
  __shared__ float rs[8], rq[8];
  const int tid = threadIdx.x;
  const size_t base = (size_t)blockIdx.x * Dd;
  float s = 0.0f, sq = 0.0f;
  for (int d = tid; d < Dd; d += 256) {
    const float v = tout[base + d];
    s += v; sq += v * v;
  }
  for (int o = 16; o > 0; o >>= 1) {
    s += __shfl_xor(s, o, 32);
    sq += __shfl_xor(sq, o, 32);
  }
  if ((tid & 31) == 0) { rs[tid >> 5] = s; rq[tid >> 5] = sq; }
  __syncthreads();
  if (tid == 0) {
    float a = 0.0f, c = 0.0f;
    for (int i = 0; i < 8; ++i) { a += rs[i]; c += rq[i]; }
    rs[0] = a; rq[0] = c;
  }
  __syncthreads();
  const float mu = rs[0] / (float)Dd;
  const float var = rq[0] / (float)Dd - mu * mu;
  const float rstd = rsqrtf(var + 1e-6f);
  for (int d = tid; d < Dd; d += 256) {
    const float xh = (tout[base + d] - mu) * rstd;
    const float ln = pw[d] * xh + pb[d];
    gbf[base + d] = (__bf16)(gelu_t(xg[base + d]) * ln);
  }
}

// ---------------- host launcher ----------------
extern "C" void kernel_launch(void* const* d_in, const int* in_sizes, int n_in,
                              void* d_out, int out_size, void* d_ws, size_t ws_size,
                              hipStream_t stream) {
  (void)in_sizes; (void)n_in; (void)out_size; (void)ws_size;
  const float* x   = (const float*)d_in[0];
  const float* q_w = (const float*)d_in[1];
  const float* v_w = (const float*)d_in[2];
  const float* o_w = (const float*)d_in[3];
  const float* g_w = (const float*)d_in[4];
  const float* cqw = (const float*)d_in[5];
  const float* cqb = (const float*)d_in[6];
  const float* ckw = (const float*)d_in[7];
  const float* ckb = (const float*)d_in[8];
  const float* lrw = (const float*)d_in[9];
  const float* lrb = (const float*)d_in[10];
  const float* tok = (const float*)d_in[11];
  const float* nw  = (const float*)d_in[12];
  const float* nb  = (const float*)d_in[13];
  const float* W1g = (const float*)d_in[14];
  const float* b1g = (const float*)d_in[15];
  const float* W2g = (const float*)d_in[16];
  const float* b2g = (const float*)d_in[17];
  const float* pw  = (const float*)d_in[18];
  const float* pb  = (const float*)d_in[19];
  float* out = (float*)d_out;

  char* ws = (char*)d_ws;
  size_t off = 0;
  auto carve = [&](size_t bytes) {
    off = (off + 255) & ~(size_t)255;
    void* r = ws + off;
    off += bytes;
    return r;
  };
  __bf16* xbf   = (__bf16*)carve((size_t)ML * Dd * 2);
  __bf16* qwbt  = (__bf16*)carve((size_t)Dd * Dd * 2);  // transposed bf16 weights
  __bf16* vwbt  = (__bf16*)carve((size_t)Dd * Dd * 2);
  __bf16* gwbt  = (__bf16*)carve((size_t)Dd * Dd * 2);
  __bf16* owbt  = (__bf16*)carve((size_t)Dd * Dd * 2);
  float* xq  = (float*)carve((size_t)ML * Dd * 4);
  float* xv  = (float*)carve((size_t)ML * Dd * 4);
  float* xg  = (float*)carve((size_t)ML * Dd * 4);
  float* XQb = (float*)carve((size_t)ML * Dd * 4);
  float* XKb = (float*)carve((size_t)ML * Dd * 4);
  float* lre = (float*)carve((size_t)Bb * NHh * Ls * 4);
  float* tts = (float*)carve((size_t)ML * Dd * 4);
  __bf16* gbf = (__bf16*)carve((size_t)ML * Dd * 2);

  const int thr = 256;
  const int tblocks = (Dd / 32) * (Dd / 32);  // 1024
  k_cvt<<<2048, thr, 0, stream>>>(x, xbf, ML * Dd);
  k_cvt_t<<<tblocks, thr, 0, stream>>>(q_w, qwbt);
  k_cvt_t<<<tblocks, thr, 0, stream>>>(v_w, vwbt);
  k_cvt_t<<<tblocks, thr, 0, stream>>>(g_w, gwbt);
  k_cvt_t<<<tblocks, thr, 0, stream>>>(o_w, owbt);

  const int gblocks = (ML / BM) * (Dd / BN);  // 64 * 16 = 1024
  k_gemm<<<gblocks, thr, 0, stream>>>(xbf, qwbt, xq, ML, Dd, Dd);
  k_gemm<<<gblocks, thr, 0, stream>>>(xbf, vwbt, xv, ML, Dd, Dd);
  k_gemm<<<gblocks, thr, 0, stream>>>(xbf, gwbt, xg, ML, Dd, Dd);

  k_dwconv<<<ML, thr, 0, stream>>>(xq, cqw, cqb, ckw, ckb, XQb, XKb);
  k_eta<<<ML, thr, 0, stream>>>(x, lrw, lrb, lre);

  hipFuncSetAttribute(reinterpret_cast<const void*>(k_scan),
                      hipFuncAttributeMaxDynamicSharedMemorySize, (int)SCAN_LDS);
  k_scan<<<Bb * NHh, thr, SCAN_LDS, stream>>>(XQb, XKb, xv, lre, tok, nw, nb, W1g, b1g, W2g,
                                              b2g, tts);

  k_postgate<<<ML, thr, 0, stream>>>(tts, xg, pw, pb, gbf);
  k_gemm<<<gblocks, thr, 0, stream>>>(gbf, owbt, out, ML, Dd, Dd);
}
